// Cross_Attention_31233002176891
// MI455X (gfx1250) — matrix operand order repaired and checked by reading, not compiled, not run on hardware
//
#include <hip/hip_runtime.h>
#include <hip/hip_bf16.h>

// ---------------------------------------------------------------------------
// Cross attention, fused per (b,n) site.  B=8 N=2048 I=64 J=32 DA=64 DB=16.
// One 256-thread block (8 wave32 waves) per site; all GEMMs via
// v_wmma_f32_16x16x32_f16; softmax/residual/LayerNorm in f32.
// HBM-bound (~544 MB @ 23.3 TB/s): all bulk global traffic is b128.
// ---------------------------------------------------------------------------

typedef __attribute__((ext_vector_type(16))) _Float16 v16h;
typedef __attribute__((ext_vector_type(8)))  _Float16 v8h;
typedef __attribute__((ext_vector_type(4)))  _Float16 v4h;
typedef __attribute__((ext_vector_type(8)))  float    v8f;

#define NSITES (8 * 2048)
#define SCALE  0.125f   // 64^-0.5

// ---- LDS layout (bytes), regions reused across barriers --------------------
#define OFF_BUF0 0          // Xh [64][64] f16 (8192)  -> later Oh [64][64] f16
#define OFF_XF   8192       // Xf [64][64] f32 (16384)           (residual)
#define OFF_WQ   24576      // WqT [64][64] f16 (8192) -> later Zf low half
#define OFF_QH   32768      // Qh  [64][64] f16 (8192) -> later Zf high half
#define OFF_ZF   24576      // Zf  [64][64] f32 (16384) overlays WqT+Qh
#define OFF_TH   40960      // Th  [32][32] f16 (2048, K padded)
#define OFF_WK   43008      // WkT [64][32] f16 (4096, K padded)
#define OFF_WV   47104      // WvT [64][32] f16 (4096, K padded)
#define OFF_SF   40960      // Sf  [64][32] f32 (8192) overlays Th/WkT
#define OFF_WO   40960      // WoT [64][64] f16 (8192) overlays Sf (after softmax)
#define OFF_KM   51200      // Km  [32][64] f16 (4096) -> later Ph [64][32] f16
#define OFF_PH   51200
#define OFF_VT   55296      // Vt  [64][32] f16 (4096)   (V transposed)
#define OFF_BO   59392      // bo   [64] f32 (256)
#define OFF_LG   59648      // ln_g [64] f32 (256)
#define OFF_LB   59904      // ln_b [64] f32 (256)
#define OFF_MU   60160      // row mean [64] f32 (256)
#define OFF_RS   60416      // row rstd [64] f32 (256)
#define SMEM_BYTES 60672

// ---- WMMA helpers (CDNA5 16x16x32 f16 -> f32) ------------------------------
__device__ __forceinline__ v8f wmma_f16(v16h a, v16h b, v8f c) {
    // (neg_a, A, neg_b, B, c_mod, C, reuse_a, reuse_b)
    return __builtin_amdgcn_wmma_f32_16x16x32_f16(false, a, false, b,
                                                  (short)0, c, false, false);
}

// A fragment, 16-bit A 16x32 (MxK) per ISA layout:
//   lane l: m = m0 + (l&15), hi = l>>4
//   halves 0..7  -> K = k0 + hi*8  + h      (contiguous run of 8)
//   halves 8..15 -> K = k0 + 16 + hi*8 + h  (contiguous run of 8)
__device__ __forceinline__ v16h load_frag_a(const _Float16* base, int lda,
                                            int m0, int k0, int lane) {
    int m  = m0 + (lane & 15);
    int hi = lane >> 4;
    const _Float16* p = base + m * lda + k0 + hi * 8;
    union { v16h v; v8h h[2]; } u;
    u.h[0] = *(const v8h*)(p);
    u.h[1] = *(const v8h*)(p + 16);
    return u.v;
}

// B fragment, 16-bit B 32x16 (KxN); baseT is stored [N][K] (row stride ldk):
//   lane l: n = n0 + (l&15), hi = l>>4; halves 0..7 -> K = k0 + hi*8 + h,
//   halves 8..15 -> K = k0 + 16 + hi*8 + (h-8)
__device__ __forceinline__ v16h load_frag_b(const _Float16* baseT, int ldk,
                                            int n0, int k0, int lane) {
    int n  = n0 + (lane & 15);
    int hi = lane >> 4;
    const _Float16* p = baseT + n * ldk + k0 + hi * 8;
    union { v16h v; v8h h[2]; } u;
    u.h[0] = *(const v8h*)(p);
    u.h[1] = *(const v8h*)(p + 16);
    return u.v;
}

// D fragment store: lane l holds column n = n0+(l&15), rows m0 + 8*(l>>4) + e
__device__ __forceinline__ void store_d_f16(_Float16* base, int ldd,
                                            int m0, int n0, int lane, v8f d) {
    int n = n0 + (lane & 15);
    int m = m0 + (lane >> 4) * 8;
#pragma unroll
    for (int e = 0; e < 8; ++e) base[(m + e) * ldd + n] = (_Float16)d[e];
}

__device__ __forceinline__ void store_d_f32(float* base, int ldd,
                                            int m0, int n0, int lane, v8f d) {
    int n = n0 + (lane & 15);
    int m = m0 + (lane >> 4) * 8;
#pragma unroll
    for (int e = 0; e < 8; ++e) base[(m + e) * ldd + n] = d[e];
}

// Transposed store (for Vt): dst[n][m..m+7] is contiguous -> single 16B store
__device__ __forceinline__ void store_d_f16_T(_Float16* base, int ldd,
                                              int m0, int n0, int lane, v8f d) {
    int n = n0 + (lane & 15);
    int m = m0 + (lane >> 4) * 8;
    v8h h;
#pragma unroll
    for (int e = 0; e < 8; ++e) h[e] = (_Float16)d[e];
    *(v8h*)(base + n * ldd + m) = h;
}

__global__ __launch_bounds__(256) void cross_attn_fused_kernel(
    const float* __restrict__ Xg_all,   // [S,64,64]
    const float* __restrict__ Tg_all,   // [S,32,16]
    const float* __restrict__ Wq,       // [64,64] (in,out)
    const float* __restrict__ Wk,       // [16,64]
    const float* __restrict__ Wv,       // [16,64]
    const float* __restrict__ Wo,       // [64,64]
    const float* __restrict__ bo,       // [64]
    const float* __restrict__ ln_g,     // [64]
    const float* __restrict__ ln_b,     // [64]
    float* __restrict__ out)            // [S,64,64]
{
    __shared__ __align__(16) char smem[SMEM_BYTES];

    const int site = blockIdx.x;
    const int tid  = threadIdx.x;
    const int lane = tid & 31;
    const int wave = tid >> 5;

    _Float16* Xh  = (_Float16*)(smem + OFF_BUF0);
    float*    Xf  = (float*)   (smem + OFF_XF);
    _Float16* WqT = (_Float16*)(smem + OFF_WQ);
    _Float16* Qh  = (_Float16*)(smem + OFF_QH);
    _Float16* Th  = (_Float16*)(smem + OFF_TH);
    _Float16* WkT = (_Float16*)(smem + OFF_WK);
    _Float16* WvT = (_Float16*)(smem + OFF_WV);
    _Float16* Km  = (_Float16*)(smem + OFF_KM);
    _Float16* Vt  = (_Float16*)(smem + OFF_VT);
    float*    Sf  = (float*)   (smem + OFF_SF);
    _Float16* Ph  = (_Float16*)(smem + OFF_PH);
    _Float16* Oh  = (_Float16*)(smem + OFF_BUF0);
    _Float16* WoT = (_Float16*)(smem + OFF_WO);
    float*    Zf  = (float*)   (smem + OFF_ZF);
    float*    bos = (float*)   (smem + OFF_BO);
    float*    lgs = (float*)   (smem + OFF_LG);
    float*    lbs = (float*)   (smem + OFF_LB);
    float*    mus = (float*)   (smem + OFF_MU);
    float*    rss = (float*)   (smem + OFF_RS);

    const float* Xg = Xg_all + (size_t)site * 64 * 64;
    const float* Tg = Tg_all + (size_t)site * 32 * 16;

    // ---- Stage 1: stage activations + weights into LDS (f16) ---------------
    // X: 4096 f32 = 1024 float4 -> global_load_b128, keep f32 + f16 copies
    {
        const float4* X4 = (const float4*)Xg;
#pragma unroll
        for (int it = 0; it < 4; ++it) {
            int i4 = tid + it * 256;
            float4 x = X4[i4];
            *(float4*)(Xf + i4 * 4) = x;
            v4h h;
            h[0] = (_Float16)x.x; h[1] = (_Float16)x.y;
            h[2] = (_Float16)x.z; h[3] = (_Float16)x.w;
            *(v4h*)(Xh + i4 * 4) = h;
        }
    }
    // T padded [32][32] (coalesced b32 reads; each LDS half written once)
    for (int i = tid; i < 1024; i += 256) {
        int j = i >> 5, k = i & 31;
        Th[i] = (_Float16)((k < 16) ? Tg[j * 16 + k] : 0.0f);
    }
    // WqT[n][k] = Wq[k][n] : b128 global reads, transposed b16 LDS writes
    {
        const float4* Wq4 = (const float4*)Wq;
#pragma unroll
        for (int it = 0; it < 4; ++it) {
            int i4 = tid + it * 256;               // 1024 float4
            int k = i4 >> 4, n0 = (i4 & 15) * 4;
            float4 w = Wq4[i4];
            WqT[(n0 + 0) * 64 + k] = (_Float16)w.x;
            WqT[(n0 + 1) * 64 + k] = (_Float16)w.y;
            WqT[(n0 + 2) * 64 + k] = (_Float16)w.z;
            WqT[(n0 + 3) * 64 + k] = (_Float16)w.w;
        }
    }
    // WkT/WvT [64][32], K padded to 32 with zeros
    for (int i = tid; i < 2048; i += 256) {
        int n = i >> 5, k = i & 31;
        WkT[i] = (_Float16)((k < 16) ? Wk[k * 64 + n] : 0.0f);
        WvT[i] = (_Float16)((k < 16) ? Wv[k * 64 + n] : 0.0f);
    }
    if (tid < 64) { bos[tid] = bo[tid]; lgs[tid] = ln_g[tid]; lbs[tid] = ln_b[tid]; }
    __syncthreads();

    // ---- Stage 2: Q = X@Wq (16 tiles, K=64); K,V = T@W (8 tiles each, K=32) -
#pragma unroll
    for (int half = 0; half < 2; ++half) {         // Q: wave w -> tiles w, w+8
        int t  = wave + half * 8;
        int mt = t >> 2, nt = t & 3;
        v8f acc = {};
#pragma unroll
        for (int kt = 0; kt < 2; ++kt) {
            v16h a = load_frag_a(Xh, 64, mt * 16, kt * 32, lane);
            v16h b = load_frag_b(WqT, 64, nt * 16, kt * 32, lane);
            acc = wmma_f16(a, b, acc);
        }
        store_d_f16(Qh, 64, mt * 16, nt * 16, lane, acc);
    }
    {
        const _Float16* Wt = (wave < 4) ? WkT : WvT;
        int wv = wave & 3;
#pragma unroll
        for (int tt = 0; tt < 2; ++tt) {
            int t  = wv * 2 + tt;                  // 0..7 over [2 x 4] tiles
            int mt = t & 1, nt = t >> 1;           // D is [32 x 64]
            v16h a = load_frag_a(Th, 32, mt * 16, 0, lane);
            v16h b = load_frag_b(Wt, 32, nt * 16, 0, lane);
            v8f acc = {};
            acc = wmma_f16(a, b, acc);
            if (wave < 4) store_d_f16 (Km, 64, mt * 16, nt * 16, lane, acc);
            else          store_d_f16_T(Vt, 32, mt * 16, nt * 16, lane, acc);
        }
    }
    __syncthreads();

    // ---- Stage 3: S = Q @ K^T  (D [64x32], 8 tiles, K=64) ------------------
    {
        int mt = wave >> 1, nt = wave & 1;
        v8f acc = {};
#pragma unroll
        for (int kt = 0; kt < 2; ++kt) {
            v16h a = load_frag_a(Qh, 64, mt * 16, kt * 32, lane);
            v16h b = load_frag_b(Km, 64, nt * 16, kt * 32, lane); // Km rows = j
            acc = wmma_f16(a, b, acc);
        }
        store_d_f32(Sf, 32, mt * 16, nt * 16, lane, acc);
    }
    __syncthreads();

    // ---- Stage 4: row softmax (scale 1/8) -> Ph (f16) ----------------------
    if (tid < 64) {
        int r = tid;
        float s[32];
        float m = -1e30f;
#pragma unroll
        for (int j = 0; j < 32; ++j) {
            float v = Sf[r * 32 + j] * SCALE;
            s[j] = v;
            m = fmaxf(m, v);
        }
        float sum = 0.0f;
#pragma unroll
        for (int j = 0; j < 32; ++j) { float e = __expf(s[j] - m); s[j] = e; sum += e; }
        float inv = 1.0f / sum;
#pragma unroll
        for (int j = 0; j < 32; ++j) Ph[r * 32 + j] = (_Float16)(s[j] * inv);
    }
    __syncthreads();

    // ---- Stage 5: WoT load (reuses Sf region)  +  O = P@V + X --------------
    {
        const float4* Wo4 = (const float4*)Wo;
#pragma unroll
        for (int it = 0; it < 4; ++it) {
            int i4 = tid + it * 256;               // WoT[n][k] = Wo[k][n]
            int k = i4 >> 4, n0 = (i4 & 15) * 4;
            float4 w = Wo4[i4];
            WoT[(n0 + 0) * 64 + k] = (_Float16)w.x;
            WoT[(n0 + 1) * 64 + k] = (_Float16)w.y;
            WoT[(n0 + 2) * 64 + k] = (_Float16)w.z;
            WoT[(n0 + 3) * 64 + k] = (_Float16)w.w;
        }
    }
#pragma unroll
    for (int half = 0; half < 2; ++half) {         // O: 16 tiles, K=32
        int t  = wave + half * 8;
        int mt = t >> 2, nt = t & 3;
        v16h a = load_frag_a(Ph, 32, mt * 16, 0, lane);
        v16h b = load_frag_b(Vt, 32, nt * 16, 0, lane);
        v8f acc = {};
        acc = wmma_f16(a, b, acc);
        int n  = nt * 16 + (lane & 15);
        int m0 = mt * 16 + (lane >> 4) * 8;
#pragma unroll
        for (int e = 0; e < 8; ++e) {
            float o = acc[e] + Xf[(m0 + e) * 64 + n];     // residual (f32)
            Oh[(m0 + e) * 64 + n] = (_Float16)o;          // Oh overlays dead Xh
        }
    }
    __syncthreads();

    // ---- Stage 6: Z = O @ Wo + bo  (f32, overlays dead WqT/Qh) -------------
#pragma unroll
    for (int half = 0; half < 2; ++half) {
        int t  = wave + half * 8;
        int mt = t >> 2, nt = t & 3;
        v8f acc = {};
#pragma unroll
        for (int kt = 0; kt < 2; ++kt) {
            v16h a = load_frag_a(Oh, 64, mt * 16, kt * 32, lane);
            v16h b = load_frag_b(WoT, 64, nt * 16, kt * 32, lane);
            acc = wmma_f16(a, b, acc);
        }
        int n  = nt * 16 + (lane & 15);
        int m0 = mt * 16 + (lane >> 4) * 8;
        float bias = bos[n];
#pragma unroll
        for (int e = 0; e < 8; ++e) Zf[(m0 + e) * 64 + n] = acc[e] + bias;
    }
    __syncthreads();

    // ---- Stage 7: LayerNorm stats per row ----------------------------------
    if (tid < 64) {
        int r = tid;
        const float* z = Zf + r * 64;
        float mu = 0.0f;
#pragma unroll
        for (int c = 0; c < 64; ++c) mu += z[c];
        mu *= (1.0f / 64.0f);
        float var = 0.0f;
#pragma unroll
        for (int c = 0; c < 64; ++c) { float d = z[c] - mu; var += d * d; }
        var *= (1.0f / 64.0f);
        mus[r] = mu;
        rss[r] = rsqrtf(var + 1e-5f);
    }
    __syncthreads();

    // ---- Stage 8: normalized write-out, b128 stores -------------------------
    {
        float4* og4 = (float4*)(out + (size_t)site * 4096);
#pragma unroll
        for (int it = 0; it < 4; ++it) {
            int i4 = tid + it * 256;               // 1024 float4
            int r = i4 >> 4, c0 = (i4 & 15) * 4;
            float4 z = *(const float4*)(Zf + i4 * 4);
            float mu = mus[r], rs = rss[r];
            float4 o;
            o.x = (z.x - mu) * rs * lgs[c0 + 0] + lbs[c0 + 0];
            o.y = (z.y - mu) * rs * lgs[c0 + 1] + lbs[c0 + 1];
            o.z = (z.z - mu) * rs * lgs[c0 + 2] + lbs[c0 + 2];
            o.w = (z.w - mu) * rs * lgs[c0 + 3] + lbs[c0 + 3];
            og4[i4] = o;
        }
    }
}

extern "C" void kernel_launch(void* const* d_in, const int* in_sizes, int n_in,
                              void* d_out, int out_size, void* d_ws, size_t ws_size,
                              hipStream_t stream) {
    (void)in_sizes; (void)n_in; (void)out_size; (void)d_ws; (void)ws_size;
    const float* X    = (const float*)d_in[0];  // voxel_code [8,2048,64,64]
    const float* T    = (const float*)d_in[1];  // trans_code [8,2048,32,16]
    const float* Wq   = (const float*)d_in[2];
    const float* Wk   = (const float*)d_in[3];
    const float* Wv   = (const float*)d_in[4];
    const float* Wo   = (const float*)d_in[5];
    const float* bo   = (const float*)d_in[6];
    const float* ln_g = (const float*)d_in[7];
    const float* ln_b = (const float*)d_in[8];
    float* out = (float*)d_out;

    cross_attn_fused_kernel<<<dim3(NSITES), dim3(256), 0, stream>>>(
        X, T, Wq, Wk, Wv, Wo, bo, ln_g, ln_b, out);
}
